// MultiheadAttention_40948218200805
// MI455X (gfx1250) — matrix operand order repaired and checked by reading, not compiled, not run on hardware
//
#include <hip/hip_runtime.h>
#include <hip/hip_bf16.h>

// -------- types / WMMA helpers (gfx1250 wave32, 16x16x32 f16 WMMA) --------
typedef __attribute__((ext_vector_type(16))) _Float16 v16h;
typedef __attribute__((ext_vector_type(8)))  _Float16 v8h;
typedef __attribute__((ext_vector_type(8)))  float    v8f;

union H16 { v16h v; v8h h[2]; };

// A-fragment: lane row = lane&15 ; 16 contiguous K values
__device__ __forceinline__ v16h frag_a(const _Float16* p) {
  H16 u; u.h[0] = *(const v8h*)p; u.h[1] = *(const v8h*)(p + 8); return u.v;
}
// B-fragment: lane col = lane&15 ; 16 contiguous K values
__device__ __forceinline__ v16h frag_b(const _Float16* p) {
  H16 u; u.h[0] = *(const v8h*)p; u.h[1] = *(const v8h*)(p + 8); return u.v;
}
__device__ __forceinline__ v8f wmma16(v16h a, v16h b, v8f c) {
  // (neg_a, A, neg_b, B, c_mod, C, reuse_a, reuse_b)
  return __builtin_amdgcn_wmma_f32_16x16x32_f16(false, a, false, b, (short)0, c, false, false);
}
__device__ __forceinline__ v8f zero8() {
  v8f z;
#pragma unroll
  for (int i = 0; i < 8; ++i) z[i] = 0.0f;
  return z;
}
__device__ __forceinline__ void lds_wave_fence() {
  asm volatile("s_wait_dscnt 0" ::: "memory");
}

// -------- constants (from reference shapes) --------
#define TSEQ 2048
#define SSEQ 2048
#define BSZ  4
#define NH   8
#define DH   64
#define HID  256
#define NROW (TSEQ * BSZ)   // 8192

// ---------------- weight transpose + f32->f16 convert ----------------
// in: [R][C] f32 row-major  ->  out: [C][R] f16 (K-contiguous for B-fragments)
__global__ void cvt_t_kernel(const float* __restrict__ in, _Float16* __restrict__ out,
                             int R, int C) {
  int idx = blockIdx.x * blockDim.x + threadIdx.x;
  if (idx < R * C) {
    int r = idx / C, c = idx - r * C;
    out[(size_t)c * R + r] = (_Float16)in[idx];
  }
}

// ---------------- fused 2-layer MLP projection ----------------
// X [8192][64] f32 -> Y = relu(X@W1+b1)@W2+b2 (f16), written per-head.
// mode 0: q  -> [B*H][T][64], scaled by 0.125
// mode 1: k  -> [B*H][S][64]
// mode 2: v  -> [B*H][64][S]   (transposed, so PV B-frags are K-contiguous)
__global__ __launch_bounds__(256)
void mlp_kernel(const float* __restrict__ X, const _Float16* __restrict__ W1t,
                const float* __restrict__ b1, const _Float16* __restrict__ W2t,
                const float* __restrict__ b2, _Float16* __restrict__ dst, int mode)
{
  __shared__ __align__(16) _Float16 Xs[64 * 72];    // 64 rows x 64 k (stride 72)
  __shared__ __align__(16) _Float16 Hs[64 * 264];   // 64 rows x 256 k (stride 264)
  const int tid = threadIdx.x;
  const int lane = tid & 31, w = tid >> 5;
  const int lr = lane & 15, hi = lane >> 4;
  const int n0 = blockIdx.x * 64;

  { // load X tile, convert f32 -> f16
    int row  = tid >> 2;
    int part = (tid & 3) * 16;
    const float* src = X + (size_t)(n0 + row) * 64 + part;
    _Float16 tmp[16];
#pragma unroll
    for (int i = 0; i < 16; ++i) tmp[i] = (_Float16)src[i];
    v8h* d = (v8h*)&Xs[row * 72 + part];
    d[0] = *(v8h*)&tmp[0];
    d[1] = *(v8h*)&tmp[8];
  }
  __syncthreads();

  // stage A: H = relu(X @ W1 + b1)   (64x256)
  for (int id = w; id < 64; id += 8) {
    int r = id & 3, c = id >> 2;
    v8f acc = zero8();
#pragma unroll
    for (int kk = 0; kk < 2; ++kk) {
      v16h a = frag_a(&Xs[(r * 16 + lr) * 72 + kk * 32 + hi * 16]);
      v16h b = frag_b(&W1t[(size_t)(c * 16 + lr) * 64 + kk * 32 + hi * 16]);
      acc = wmma16(a, b, acc);
    }
    float bias = b1[c * 16 + lr];
#pragma unroll
    for (int i = 0; i < 8; ++i) {
      float hv = acc[i] + bias;
      hv = hv > 0.0f ? hv : 0.0f;
      Hs[(r * 16 + i + 8 * hi) * 264 + c * 16 + lr] = (_Float16)hv;
    }
  }
  __syncthreads();

  // stage B: Y = H @ W2 + b2   (64x512)
  const float scale = (mode == 0) ? 0.125f : 1.0f;
  for (int id = w; id < 128; id += 8) {
    int r = id & 3, c = id >> 2;           // c: 0..31
    v8f acc = zero8();
#pragma unroll
    for (int kk = 0; kk < 8; ++kk) {
      v16h a = frag_a(&Hs[(r * 16 + lr) * 264 + kk * 32 + hi * 16]);
      v16h b = frag_b(&W2t[(size_t)(c * 16 + lr) * 256 + kk * 32 + hi * 16]);
      acc = wmma16(a, b, acc);
    }
    float bias = b2[c * 16 + lr];
    int outc = c * 16 + lr;
    int hidx = outc >> 6, d = outc & 63;
#pragma unroll
    for (int i = 0; i < 8; ++i) {
      int n = n0 + r * 16 + i + 8 * hi;
      int t = n >> 2, bi = n & 3;          // n = t*B + b
      float y = (acc[i] + bias) * scale;
      size_t addr;
      if (mode == 2) addr = ((size_t)(bi * NH + hidx) * DH + d) * (size_t)SSEQ + t;
      else           addr = ((size_t)(bi * NH + hidx) * (size_t)TSEQ + t) * DH + d;
      dst[addr] = (_Float16)y;
    }
  }
}

// ---------------- flash-style attention ----------------
// grid.x = T/128, grid.y = B*H ; 8 waves, each owns a 16-row Q strip.
__global__ __launch_bounds__(256)
void attn_kernel(const _Float16* __restrict__ Q, const _Float16* __restrict__ K,
                 const _Float16* __restrict__ Vt, const float* __restrict__ mask,
                 _Float16* __restrict__ O)
{
  __shared__ __align__(16) _Float16 Ks[64 * 72];       // [s][d]
  __shared__ __align__(16) _Float16 Vs[64 * 72];       // [d][s]
  __shared__ __align__(16) _Float16 Ps[8 * 16 * 72];   // per-wave P transpose buffer
  const int tid = threadIdx.x;
  const int lane = tid & 31, w = tid >> 5;
  const int lr = lane & 15, hi = lane >> 4;
  const int bh = blockIdx.y;
  const int bidx = bh >> 3, h = bh & 7;
  const int tw = blockIdx.x * 128 + w * 16;
  _Float16* Psw = &Ps[w * 16 * 72];

  // persistent Q fragments (A-matrix, K = d = 64 -> 2 frags)
  const _Float16* qb = Q + ((size_t)bh * TSEQ + tw + lr) * DH + hi * 16;
  v16h aq0 = frag_a(qb);
  v16h aq1 = frag_a(qb + 32);

  float m[8], l[8];
  v8f o[4];
#pragma unroll
  for (int i = 0; i < 8; ++i) { m[i] = -3.0e30f; l[i] = 0.0f; }
#pragma unroll
  for (int dt = 0; dt < 4; ++dt) o[dt] = zero8();

  const float* maskb = mask + ((size_t)bidx * TSEQ + tw + 8 * hi) * (size_t)SSEQ + lr;

  for (int s0 = 0; s0 < SSEQ; s0 += 64) {
    { // cooperative K / V^T tile load into LDS
      int row  = tid >> 2;
      int part = (tid & 3) * 16;
      const _Float16* ksrc = K  + ((size_t)bh * SSEQ + s0 + row) * DH + part;
      v8h* kd = (v8h*)&Ks[row * 72 + part];
      kd[0] = *(const v8h*)ksrc; kd[1] = *(const v8h*)(ksrc + 8);
      const _Float16* vsrc = Vt + ((size_t)bh * DH + row) * (size_t)SSEQ + s0 + part;
      v8h* vd = (v8h*)&Vs[row * 72 + part];
      vd[0] = *(const v8h*)vsrc; vd[1] = *(const v8h*)(vsrc + 8);
    }
    __syncthreads();

    // scores = q . k^T + mask   (16 x 64 per wave)
    v8f sc[4];
#pragma unroll
    for (int st = 0; st < 4; ++st) {
      sc[st] = zero8();
      v16h b0 = frag_b(&Ks[(st * 16 + lr) * 72 + hi * 16]);
      sc[st] = wmma16(aq0, b0, sc[st]);
      v16h b1 = frag_b(&Ks[(st * 16 + lr) * 72 + 32 + hi * 16]);
      sc[st] = wmma16(aq1, b1, sc[st]);
      const float* mp = maskb + s0 + st * 16;
#pragma unroll
      for (int i = 0; i < 8; ++i) sc[st][i] += mp[(size_t)i * SSEQ];
    }

    // online softmax (rows are spread over 16 lanes in C/D layout)
    float mnew[8], alpha[8], rsum[8];
#pragma unroll
    for (int i = 0; i < 8; ++i) {
      float r = sc[0][i];
      r = fmaxf(r, sc[1][i]); r = fmaxf(r, sc[2][i]); r = fmaxf(r, sc[3][i]);
      for (int x = 1; x < 16; x <<= 1) r = fmaxf(r, __shfl_xor(r, x, 32));
      mnew[i]  = fmaxf(m[i], r);
      alpha[i] = __expf(m[i] - mnew[i]);
      m[i]     = mnew[i];
      rsum[i]  = 0.0f;
    }
#pragma unroll
    for (int st = 0; st < 4; ++st)
#pragma unroll
      for (int i = 0; i < 8; ++i) {
        float p = __expf(sc[st][i] - mnew[i]);
        sc[st][i] = p;
        rsum[i] += p;
      }
#pragma unroll
    for (int i = 0; i < 8; ++i) {
      float r = rsum[i];
      for (int x = 1; x < 16; x <<= 1) r += __shfl_xor(r, x, 32);
      l[i] = l[i] * alpha[i] + r;
    }
#pragma unroll
    for (int dt = 0; dt < 4; ++dt)
#pragma unroll
      for (int i = 0; i < 8; ++i) o[dt][i] *= alpha[i];

    // transpose P through wave-private LDS into A-fragment layout
#pragma unroll
    for (int st = 0; st < 4; ++st)
#pragma unroll
      for (int i = 0; i < 8; ++i)
        Psw[(i + 8 * hi) * 72 + st * 16 + lr] = (_Float16)sc[st][i];
    lds_wave_fence();
    v16h pa0 = frag_a(&Psw[lr * 72 + hi * 16]);
    v16h pa1 = frag_a(&Psw[lr * 72 + 32 + hi * 16]);

    // O += P @ V
#pragma unroll
    for (int dt = 0; dt < 4; ++dt) {
      v16h b0 = frag_b(&Vs[(dt * 16 + lr) * 72 + hi * 16]);
      o[dt] = wmma16(pa0, b0, o[dt]);
      v16h b1 = frag_b(&Vs[(dt * 16 + lr) * 72 + 32 + hi * 16]);
      o[dt] = wmma16(pa1, b1, o[dt]);
    }
    __syncthreads();
  }

  // epilogue: normalize, write [8192][512] f16 (row n = t*B+b, col = h*64+d)
#pragma unroll
  for (int dt = 0; dt < 4; ++dt) {
    int d = h * DH + dt * 16 + lr;
#pragma unroll
    for (int i = 0; i < 8; ++i) {
      int t = tw + i + 8 * hi;
      float y = o[dt][i] / l[i];
      O[((size_t)t * BSZ + bidx) * (size_t)(NH * DH) + d] = (_Float16)y;
    }
  }
}

// ---------------- output projection: out = AO @ Wo + bo ----------------
__global__ __launch_bounds__(256)
void oproj_kernel(const _Float16* __restrict__ AO, const _Float16* __restrict__ Wot,
                  const float* __restrict__ bo, float* __restrict__ out)
{
  const int tid = threadIdx.x;
  const int lane = tid & 31, w = tid >> 5;
  const int lr = lane & 15, hi = lane >> 4;
  const int n0 = blockIdx.x * 128;
  for (int id = w; id < 32; id += 8) {
    int r = id & 7, c = id >> 3;            // r 0..7, c 0..3
    v8f acc = zero8();
    const _Float16* ab = AO  + (size_t)(n0 + r * 16 + lr) * 512 + hi * 16;
    const _Float16* bb = Wot + (size_t)(c * 16 + lr) * 512 + hi * 16;
#pragma unroll
    for (int kk = 0; kk < 16; ++kk) {
      v16h a = frag_a(ab + kk * 32);
      v16h b = frag_b(bb + kk * 32);
      acc = wmma16(a, b, acc);
    }
    float bias = bo[c * 16 + lr];
#pragma unroll
    for (int i = 0; i < 8; ++i) {
      int n = n0 + r * 16 + i + 8 * hi;
      out[(size_t)n * 64 + c * 16 + lr] = acc[i] + bias;
    }
  }
}

// ---------------- launcher ----------------
extern "C" void kernel_launch(void* const* d_in, const int* in_sizes, int n_in,
                              void* d_out, int out_size, void* d_ws, size_t ws_size,
                              hipStream_t stream)
{
  const float* query = (const float*)d_in[0];
  const float* key   = (const float*)d_in[1];
  const float* value = (const float*)d_in[2];
  const float* mask  = (const float*)d_in[3];
  const float* Wq1 = (const float*)d_in[4];  const float* bq1 = (const float*)d_in[5];
  const float* Wq2 = (const float*)d_in[6];  const float* bq2 = (const float*)d_in[7];
  const float* Wk1 = (const float*)d_in[8];  const float* bk1 = (const float*)d_in[9];
  const float* Wk2 = (const float*)d_in[10]; const float* bk2 = (const float*)d_in[11];
  const float* Wv1 = (const float*)d_in[12]; const float* bv1 = (const float*)d_in[13];
  const float* Wv2 = (const float*)d_in[14]; const float* bv2 = (const float*)d_in[15];
  const float* Wo  = (const float*)d_in[16]; const float* bo  = (const float*)d_in[17];

  _Float16* ws   = (_Float16*)d_ws;
  _Float16* Wq1t = ws;               // [256][64]
  _Float16* Wq2t = Wq1t + 16384;     // [512][256]
  _Float16* Wk1t = Wq2t + 131072;
  _Float16* Wk2t = Wk1t + 16384;
  _Float16* Wv1t = Wk2t + 131072;
  _Float16* Wv2t = Wv1t + 16384;
  _Float16* Wot  = Wv2t + 131072;    // [64][512]
  _Float16* qf   = Wot + 32768;      // [B*H][T][64]
  _Float16* kf   = qf  + 4194304;    // [B*H][S][64]
  _Float16* vtf  = kf  + 4194304;    // [B*H][64][S]
  _Float16* aof  = vtf + 4194304;    // [8192][512]

  // 1) weight transpose/convert
  cvt_t_kernel<<<(64 * 256 + 255) / 256, 256, 0, stream>>>(Wq1, Wq1t, 64, 256);
  cvt_t_kernel<<<(256 * 512 + 255) / 256, 256, 0, stream>>>(Wq2, Wq2t, 256, 512);
  cvt_t_kernel<<<(64 * 256 + 255) / 256, 256, 0, stream>>>(Wk1, Wk1t, 64, 256);
  cvt_t_kernel<<<(256 * 512 + 255) / 256, 256, 0, stream>>>(Wk2, Wk2t, 256, 512);
  cvt_t_kernel<<<(64 * 256 + 255) / 256, 256, 0, stream>>>(Wv1, Wv1t, 64, 256);
  cvt_t_kernel<<<(256 * 512 + 255) / 256, 256, 0, stream>>>(Wv2, Wv2t, 256, 512);
  cvt_t_kernel<<<(512 * 64 + 255) / 256, 256, 0, stream>>>(Wo, Wot, 512, 64);

  // 2) q/k/v MLP projections (WMMA)
  mlp_kernel<<<NROW / 64, 256, 0, stream>>>(query, Wq1t, bq1, Wq2t, bq2, qf, 0);
  mlp_kernel<<<NROW / 64, 256, 0, stream>>>(key,   Wk1t, bk1, Wk2t, bk2, kf, 1);
  mlp_kernel<<<NROW / 64, 256, 0, stream>>>(value, Wv1t, bv1, Wv2t, bv2, vtf, 2);

  // 3) attention (flash-style, WMMA)
  attn_kernel<<<dim3(TSEQ / 128, BSZ * NH), 256, 0, stream>>>(qf, kf, vtf, mask, aof);

  // 4) output projection (WMMA)
  oproj_kernel<<<NROW / 128, 256, 0, stream>>>(aof, Wot, bo, (float*)d_out);
}
